// SelfAttention_41772851921334
// MI455X (gfx1250) — matrix operand order repaired and checked by reading, not compiled, not run on hardware
//
#include <hip/hip_runtime.h>

typedef _Float16 half_t;
typedef __attribute__((ext_vector_type(16))) _Float16 v16h;
typedef __attribute__((ext_vector_type(8)))  _Float16 v8h;
typedef __attribute__((ext_vector_type(8)))  float    v8f;

union F16Frag { v16h v; v8h h[2]; };

#define DIM    2048
#define NHEADS 16
#define HDIM   128
#define SEQ    2048
#define BATCH  2
#define MROWS  (BATCH * SEQ)   // 4096
#define LDA    40              // padded LDS row stride in halves (80B: 16B-aligned, 16 banks)

#define USE_ASYNC 1

// Async global -> LDS copy of 16 bytes per lane (CDNA5 VGLOBAL async form).
// VDST = per-lane LDS byte offset (low 32 bits of the generic shared pointer),
// VADDR = 64-bit global address. Tracked by ASYNCcnt.
__device__ __forceinline__ void copy16_to_lds(half_t* lds, const half_t* g) {
#if USE_ASYNC
    unsigned l = (unsigned)(unsigned long long)lds;
    asm volatile("global_load_async_to_lds_b128 %0, %1, off"
                 :: "v"(l), "v"(g) : "memory");
#else
    *(v8h*)lds = *(const v8h*)g;   // global_load_b128 + ds_store_b128
#endif
}

template <int N>
__device__ __forceinline__ void wait_async() {
#if USE_ASYNC
  #if __has_builtin(__builtin_amdgcn_s_wait_asynccnt)
    __builtin_amdgcn_s_wait_asynccnt(N);
  #else
    asm volatile("s_wait_asynccnt %0" :: "n"(N) : "memory");
  #endif
#endif
}

// ---------------------------------------------------------------------------
// f32 -> f16 conversion (grid-stride)
// ---------------------------------------------------------------------------
__global__ void cvt_f32_f16(const float* __restrict__ in, half_t* __restrict__ out, long n) {
    long i = (long)blockIdx.x * blockDim.x + threadIdx.x;
    long stride = (long)gridDim.x * blockDim.x;
    for (; i < n; i += stride) out[i] = (half_t)in[i];
}

// ---------------------------------------------------------------------------
// TN GEMM: C[m,n] = sum_d A[m,d] * W[n,d],  M=MROWS, N=K=DIM
// 256 threads = 8 waves; block tile 128x128; wave tile 64x32 (4x2 wmma tiles).
// Operands staged global -> LDS via async loads, double-buffered, K-step 32.
// WMMA fragments are then 128-bit ds loads.
// mode 0: store f32 -> outF[m*DIM + n]
// mode 1: fused RoPE, store f16 -> outH[((b*16+h)*SEQ + s)*HDIM + d]
// mode 2: store f16 transposed -> outH[((b*16+h)*HDIM + d)*SEQ + s]
// ---------------------------------------------------------------------------
__global__ __launch_bounds__(256)
void gemm_qkvo(const half_t* __restrict__ A, const half_t* __restrict__ W,
               float* __restrict__ outF, half_t* __restrict__ outH, int mode)
{
    __shared__ __align__(16) half_t sA[2][128 * LDA];
    __shared__ __align__(16) half_t sB[2][128 * LDA];

    const int tid  = threadIdx.x;
    const int lane = tid & 31;
    const int wave = tid >> 5;
    const int lh = lane & 15;   // row/col select within 16-group
    const int lg = lane >> 4;   // half-wave group
    const long blkM = (long)blockIdx.y * 128;
    const long blkN = (long)blockIdx.x * 128;
    const int wm = (wave >> 2) * 64;
    const int wn = (wave & 3) * 32;

    // Stage one 128x32 A tile + 128x32 B tile. 512 16B-chunks each matrix;
    // per thread: 2 A chunks + 2 B chunks (chunk c: row=c>>2, kcol=(c&3)*8).
    auto stage = [&](int buf, int k0) {
        #pragma unroll
        for (int t = 0; t < 2; ++t) {
            int c   = tid + t * 256;
            int row = c >> 2;
            int kc  = (c & 3) * 8;
            copy16_to_lds(&sA[buf][row * LDA + kc], A + (blkM + row) * (long)DIM + k0 + kc);
            copy16_to_lds(&sB[buf][row * LDA + kc], W + (blkN + row) * (long)DIM + k0 + kc);
        }
    };

    v8f acc[4][2] = {};

    stage(0, 0);
    int buf = 0;
    for (int k0 = 0; k0 < DIM; k0 += 32) {
        const bool has_next = (k0 + 32) < DIM;
        if (has_next) stage(buf ^ 1, k0 + 32);                // prefetch next stage
        if (has_next) wait_async<4>(); else wait_async<0>();  // current stage landed
        __syncthreads();

        F16Frag a[4], b[2];
        #pragma unroll
        for (int i = 0; i < 4; ++i) {
            const half_t* p = &sA[buf][(wm + i * 16 + lh) * LDA + lg * 8];
            a[i].h[0] = *(const v8h*)p;          // K = 0..7 / 8..15
            a[i].h[1] = *(const v8h*)(p + 16);   // K = 16..23 / 24..31
        }
        #pragma unroll
        for (int j = 0; j < 2; ++j) {
            const half_t* p = &sB[buf][(wn + j * 16 + lh) * LDA + lg * 8];
            b[j].h[0] = *(const v8h*)p;
            b[j].h[1] = *(const v8h*)(p + 16);
        }
        #pragma unroll
        for (int i = 0; i < 4; ++i)
            #pragma unroll
            for (int j = 0; j < 2; ++j)
                acc[i][j] = __builtin_amdgcn_wmma_f32_16x16x32_f16(
                    false, a[i].v, false, b[j].v, (short)0, acc[i][j], false, false);

        __syncthreads();   // all waves done reading buf before it is re-staged
        buf ^= 1;
    }

    // epilogue
    #pragma unroll
    for (int i = 0; i < 4; ++i) {
        #pragma unroll
        for (int j = 0; j < 2; ++j) {
            #pragma unroll
            for (int e = 0; e < 8; ++e) {
                float val = acc[i][j][e];
                long m = blkM + wm + i * 16 + lg * 8 + e;
                long n = blkN + wn + j * 16 + lh;
                if (mode == 0) {
                    outF[m * DIM + n] = val;
                } else if (mode == 1) {
                    // fused RoPE: pair (2j, 2j+1) sits in adjacent lanes
                    int s = (int)(m & (SEQ - 1));
                    int bi = (int)(m >> 11);
                    int d = (int)(n & (HDIM - 1));
                    int h = (int)(n >> 7);
                    float partner = __shfl_xor(val, 1, 32);
                    float ang = (float)s * __expf(-(float)(d & ~1) *
                                                  (9.210340371976184f / 128.0f));
                    float sn, cs;
                    __sincosf(ang, &sn, &cs);
                    float r = (d & 1) ? (partner * sn + val * cs)
                                      : (val * cs - partner * sn);
                    outH[((long)(bi * NHEADS + h) * SEQ + s) * HDIM + d] = (half_t)r;
                } else {
                    // V stored transposed [B,H,D,S] for contiguous PV B-fragments
                    int s = (int)(m & (SEQ - 1));
                    int bi = (int)(m >> 11);
                    int d = (int)(n & (HDIM - 1));
                    int h = (int)(n >> 7);
                    outH[((long)(bi * NHEADS + h) * HDIM + d) * SEQ + s] = (half_t)val;
                }
            }
        }
    }
}

// ---------------------------------------------------------------------------
// Flash-style causal attention.
// grid = (SEQ/64, BATCH*NHEADS); block = 128 threads (4 waves).
// Wave w owns q rows [q0+16w, q0+16w+16). Q(16x128) + O(16x128 f32) in regs.
// Per iteration: 32 keys -> 8 wmma (scores) + 8 wmma (PV).
// ---------------------------------------------------------------------------
__global__ __launch_bounds__(128)
void attn_fwd(const half_t* __restrict__ Q, const half_t* __restrict__ K,
              const half_t* __restrict__ Vt, half_t* __restrict__ outH)
{
    __shared__ __align__(16) half_t pbuf[4][16 * 32];
    const int lane = threadIdx.x & 31;
    const int wave = threadIdx.x >> 5;
    const int lh = lane & 15;
    const int lg = lane >> 4;
    const int bh = blockIdx.y;                 // b*16 + h
    const int q0 = blockIdx.x * 64;
    const int qw = q0 + wave * 16;
    const long base = (long)bh * SEQ * HDIM;

    F16Frag qf[4];
    #pragma unroll
    for (int c = 0; c < 4; ++c) {
        const half_t* p = Q + base + (long)(qw + lh) * HDIM + c * 32 + lg * 8;
        qf[c].h[0] = *(const v8h*)p;
        qf[c].h[1] = *(const v8h*)(p + 16);
    }

    v8f of[8] = {};
    float mrow[8], lrow[8];
    #pragma unroll
    for (int e = 0; e < 8; ++e) { mrow[e] = -1e30f; lrow[e] = 0.0f; }

    const float scale = 0.08838834764831845f;  // 1/sqrt(128)
    const int nkt = q0 / 32 + 2;               // skip fully-masked key tiles
    for (int kt = 0; kt < nkt; ++kt) {
        const int k0 = kt * 32;

        v8f sc[2] = {};
        #pragma unroll
        for (int g = 0; g < 2; ++g) {
            #pragma unroll
            for (int c = 0; c < 4; ++c) {
                F16Frag kb;
                const half_t* p = K + base + (long)(k0 + g * 16 + lh) * HDIM
                                  + c * 32 + lg * 8;
                kb.h[0] = *(const v8h*)p;
                kb.h[1] = *(const v8h*)(p + 16);
                sc[g] = __builtin_amdgcn_wmma_f32_16x16x32_f16(
                    false, qf[c].v, false, kb.v, (short)0, sc[g], false, false);
            }
        }

        float p0[8], p1[8];
        #pragma unroll
        for (int e = 0; e < 8; ++e) {
            int q = qw + lg * 8 + e;
            int c0 = k0 + lh;
            float s0 = (c0      <= q) ? sc[0][e] * scale : -1e30f;
            float s1 = (c0 + 16 <= q) ? sc[1][e] * scale : -1e30f;
            float mx = fmaxf(s0, s1);
            mx = fmaxf(mx, __shfl_xor(mx, 1, 32));
            mx = fmaxf(mx, __shfl_xor(mx, 2, 32));
            mx = fmaxf(mx, __shfl_xor(mx, 4, 32));
            mx = fmaxf(mx, __shfl_xor(mx, 8, 32));
            float mnew = fmaxf(mrow[e], mx);
            float corr = __expf(mrow[e] - mnew);
            float e0 = __expf(s0 - mnew);
            float e1 = __expf(s1 - mnew);
            float sum = e0 + e1;
            sum += __shfl_xor(sum, 1, 32);
            sum += __shfl_xor(sum, 2, 32);
            sum += __shfl_xor(sum, 4, 32);
            sum += __shfl_xor(sum, 8, 32);
            lrow[e] = lrow[e] * corr + sum;
            mrow[e] = mnew;
            p0[e] = e0; p1[e] = e1;
            #pragma unroll
            for (int j = 0; j < 8; ++j) of[j][e] *= corr;
        }

        // re-layout P (C-fragment) -> A-fragment through per-wave LDS
        #pragma unroll
        for (int e = 0; e < 8; ++e) {
            int r = lg * 8 + e;
            pbuf[wave][r * 32 + lh]      = (half_t)p0[e];
            pbuf[wave][r * 32 + lh + 16] = (half_t)p1[e];
        }
        __syncthreads();
        F16Frag pf;
        pf.h[0] = *(const v8h*)&pbuf[wave][lh * 32 + lg * 8];
        pf.h[1] = *(const v8h*)&pbuf[wave][lh * 32 + lg * 8 + 16];
        __syncthreads();

        #pragma unroll
        for (int j = 0; j < 8; ++j) {
            F16Frag vb;
            const half_t* p = Vt + base + (long)(j * 16 + lh) * SEQ + k0 + lg * 8;
            vb.h[0] = *(const v8h*)p;
            vb.h[1] = *(const v8h*)(p + 16);
            of[j] = __builtin_amdgcn_wmma_f32_16x16x32_f16(
                false, pf.v, false, vb.v, (short)0, of[j], false, false);
        }
    }

    const int b = bh >> 4, h = bh & 15;
    #pragma unroll
    for (int j = 0; j < 8; ++j) {
        #pragma unroll
        for (int e = 0; e < 8; ++e) {
            int q = qw + lg * 8 + e;
            int d = j * 16 + lh;
            float v = of[j][e] / lrow[e];
            outH[((long)(b * SEQ + q)) * DIM + h * HDIM + d] = (half_t)v;
        }
    }
}

// ---------------------------------------------------------------------------
extern "C" void kernel_launch(void* const* d_in, const int* in_sizes, int n_in,
                              void* d_out, int out_size, void* d_ws, size_t ws_size,
                              hipStream_t stream)
{
    (void)in_sizes; (void)n_in; (void)out_size; (void)ws_size;
    const float* x  = (const float*)d_in[0];
    const float* Wq = (const float*)d_in[1];
    const float* Wk = (const float*)d_in[2];
    const float* Wv = (const float*)d_in[3];
    const float* Wo = (const float*)d_in[4];
    float* out = (float*)d_out;

    char* ws = (char*)d_ws;
    half_t* xh  = (half_t*)(ws);                 // 16 MiB (reused as attn output)
    half_t* wqh = (half_t*)(ws + (16l << 20));   //  8 MiB
    half_t* wkh = (half_t*)(ws + (24l << 20));
    half_t* wvh = (half_t*)(ws + (32l << 20));
    half_t* woh = (half_t*)(ws + (40l << 20));
    half_t* Qh  = (half_t*)(ws + (48l << 20));   // 16 MiB [B,H,S,D]
    half_t* Kh  = (half_t*)(ws + (64l << 20));   // 16 MiB [B,H,S,D]
    half_t* Vt  = (half_t*)(ws + (80l << 20));   // 16 MiB [B,H,D,S]
    half_t* attnh = xh;                          // x dead after QKV projections

    const long NX = (long)MROWS * DIM;
    const long NW = (long)DIM * DIM;
    cvt_f32_f16<<<1024, 256, 0, stream>>>(x,  xh,  NX);
    cvt_f32_f16<<<1024, 256, 0, stream>>>(Wq, wqh, NW);
    cvt_f32_f16<<<1024, 256, 0, stream>>>(Wk, wkh, NW);
    cvt_f32_f16<<<1024, 256, 0, stream>>>(Wv, wvh, NW);
    cvt_f32_f16<<<1024, 256, 0, stream>>>(Wo, woh, NW);

    dim3 g(DIM / 128, MROWS / 128);  // (16, 32)
    gemm_qkvo<<<g, 256, 0, stream>>>(xh, wqh, nullptr, Qh, 1);  // Q + RoPE
    gemm_qkvo<<<g, 256, 0, stream>>>(xh, wkh, nullptr, Kh, 1);  // K + RoPE
    gemm_qkvo<<<g, 256, 0, stream>>>(xh, wvh, nullptr, Vt, 2);  // V transposed

    attn_fwd<<<dim3(SEQ / 64, BATCH * NHEADS), 128, 0, stream>>>(Qh, Kh, Vt, attnh);

    gemm_qkvo<<<g, 256, 0, stream>>>(attnh, woh, out, nullptr, 0);  // O projection
}
